// GraphLearner_63333587746987
// MI455X (gfx1250) — compile-verified, matrix operand order checked by reading, not run on hardware
//
#include <hip/hip_runtime.h>
#include <hip/hip_bf16.h>

typedef _Float16 v16h __attribute__((ext_vector_type(16)));
typedef float    v8f  __attribute__((ext_vector_type(8)));

#define LDS_USH_STRIDE 72   // [m=32][k=64] f16, 144B rows (16B aligned, conflict-free)
#define LDS_W1T_STRIDE 72   // [n=256][k=64] f16, 144B rows (16B aligned, conflict-free)
#define LDS_W2T_STRIDE 136  // [n=128][k=128] f16, 272B rows (16B aligned, conflict-free)
#define LDS_H_STRIDE   132  // [32][128] f32, 528B rows (8B aligned, conflict-free)

// ---- WMMA fragment builders (16-bit A/B layout, 16x16x32 shape) ----
// A layout: lanes 0-15 & 16-31 both hold rows M=0..15; element e maps to
//   k = (e<8 ? e : e+8) + (lane>=16 ? 8 : 0)   (per ISA 7.12.2 table)
// B assumed symmetric: lane&15 = column n, same k-per-element map.

__device__ __forceinline__ v16h frag_from_f16(const _Float16* src, int rowStride,
                                              int nBase, int kBase, int lane) {
  int nr = lane & 15;
  int off0 = (lane >= 16) ? 8 : 0;
  const _Float16* row = src + (nBase + nr) * rowStride + kBase + off0;
  v16h out;
#pragma unroll
  for (int e = 0; e < 8; ++e) out[e] = row[e];        // contiguous -> ds_load_b128
#pragma unroll
  for (int e = 0; e < 8; ++e) out[8 + e] = row[16 + e];
  return out;
}

// A fragment of h = relu(hi[i] + hj[j] + b1) built on the fly (b1 pre-folded into hiP)
__device__ __forceinline__ v16h frag_h(const float* hiRow, const float* hjBase,
                                       int rowStride, int jBase, int kBase, int lane) {
  int mr = lane & 15;
  int off0 = (lane >= 16) ? 8 : 0;
  const float* hr = hiRow + kBase + off0;
  const float* jr = hjBase + (jBase + mr) * rowStride + kBase + off0;
  v16h out;
#pragma unroll
  for (int e = 0; e < 8; ++e) out[e] = (_Float16)fmaxf(hr[e] + jr[e], 0.0f);
#pragma unroll
  for (int e = 0; e < 8; ++e) out[8 + e] = (_Float16)fmaxf(hr[16 + e] + jr[16 + e], 0.0f);
  return out;
}

__device__ __forceinline__ v8f wmma_f32(v16h a, v16h b, v8f c) {
  return __builtin_amdgcn_wmma_f32_16x16x32_f16(false, a, false, b, (short)0, c,
                                                false, false);
}

__global__ __launch_bounds__(256)
void graph_learner_kernel(const float* __restrict__ us, const float* __restrict__ W1,
                          const float* __restrict__ b1, const float* __restrict__ W2,
                          const float* __restrict__ b2, const float* __restrict__ W3,
                          const float* __restrict__ b3, float* __restrict__ out) {
  __shared__ _Float16 usH[32 * LDS_USH_STRIDE];    // us[b] as f16 [m][k]
  __shared__ _Float16 W1T[256 * LDS_W1T_STRIDE];   // [n][k]; n<128 -> hi half, n>=128 -> hj half
  __shared__ _Float16 W2T[128 * LDS_W2T_STRIDE];   // [n][k]
  __shared__ float    hiP[32 * LDS_H_STRIDE];      // hi + b1
  __shared__ float    hjP[32 * LDS_H_STRIDE];      // hj
  __shared__ float    b1s[128], b2s[128], W3s[128];

  const int tid  = threadIdx.x;
  const int lane = tid & 31;
  const int wave = tid >> 5;
  const int b    = blockIdx.x;

  // ---- Phase 0: stage inputs into LDS (coalesced reads, transposed f16 weights) ----
  for (int idx = tid; idx < 2048; idx += 256) {            // us[b]: 32 agents x 64 -> f16
    int a = idx >> 6, d = idx & 63;
    usH[a * LDS_USH_STRIDE + d] =
        (_Float16)us[((size_t)a << 16) + ((size_t)b << 6) + d];
  }
  for (int idx = tid; idx < 16384; idx += 256) {           // W1 -> W1T f16 [256][64]
    int k = idx >> 8, n = idx & 255;
    int srcRow = k + ((n >> 7) << 6);                      // rows 0..63 / 64..127
    W1T[n * LDS_W1T_STRIDE + k] = (_Float16)W1[srcRow * 128 + (n & 127)];
  }
  for (int idx = tid; idx < 16384; idx += 256) {           // W2 -> W2T f16 [128][128]
    int k = idx >> 7, n = idx & 127;
    W2T[n * LDS_W2T_STRIDE + k] = (_Float16)W2[k * 128 + n];
  }
  if (tid < 128) { b1s[tid] = b1[tid]; b2s[tid] = b2[tid]; W3s[tid] = W3[tid]; }
  __syncthreads();

  // ---- Phase A: hi/hj = us @ W1halves  (M=32, N=256, K=64; 8 WMMAs/wave) ----
#pragma unroll
  for (int t = 0; t < 4; ++t) {
    int idx = wave * 4 + t;
    int mt = idx & 1, nt = idx >> 1;                       // 2 M-tiles x 16 N-tiles
    v8f C = {};
#pragma unroll
    for (int ks = 0; ks < 2; ++ks) {
      v16h A = frag_from_f16(usH, LDS_USH_STRIDE, mt * 16, ks * 32, lane);
      v16h B = frag_from_f16(W1T, LDS_W1T_STRIDE, nt * 16, ks * 32, lane);
      C = wmma_f32(A, B, C);
    }
    // Branch-free scatter: select destination + bias via cndmask (no exec branching)
    int nr   = lane & 15;
    int hl   = lane >> 4;
    int ncol = (nt * 16 + nr) & 127;
    bool isHi = (nt < 8);
    float* dstBase = isHi ? hiP : hjP;
    float  biasLd  = b1s[ncol];                            // unconditional broadcast load
    float  b1v     = isHi ? biasLd : 0.0f;
#pragma unroll
    for (int r = 0; r < 8; ++r) {                          // C layout: row = r / r+8
      int m = mt * 16 + r + hl * 8;
      dstBase[m * LDS_H_STRIDE + ncol] = C[r] + b1v;
    }
  }
  __syncthreads();

  const float b3v = b3[0];

  // ---- Phase B: 1024 pairs x (relu -> @W2 -> relu -> .W3 -> sigmoid) ----
  // Two M-tiles per pass: each W2 B-fragment loaded once feeds two WMMAs,
  // halving LDS read traffic per matrix op. nb loop capped at unroll 2 to
  // keep the live set under 256 VGPRs (no s_set_vgpr_msb spill machinery).
#pragma unroll 1
  for (int t = 0; t < 4; ++t) {
    int mt0 = wave + t * 16;                               // M-tiles for this wave
    int mt1 = mt0 + 8;
    int p0a = mt0 * 16, p0b = mt1 * 16;
    int ia = p0a >> 5, ib = p0b >> 5;                      // constant within each tile
    int jBaseA = p0a & 31, jBaseB = p0b & 31;              // 0 or 16
    const float* hiRowA = hiP + ia * LDS_H_STRIDE;
    const float* hiRowB = hiP + ib * LDS_H_STRIDE;

    v16h A0[4], A1[4];
#pragma unroll
    for (int ks = 0; ks < 4; ++ks) {
      A0[ks] = frag_h(hiRowA, hjP, LDS_H_STRIDE, jBaseA, ks * 32, lane);
      A1[ks] = frag_h(hiRowB, hjP, LDS_H_STRIDE, jBaseB, ks * 32, lane);
    }

    float acc0[8], acc1[8];
#pragma unroll
    for (int r = 0; r < 8; ++r) { acc0[r] = 0.0f; acc1[r] = 0.0f; }

#pragma unroll 2
    for (int nb = 0; nb < 8; ++nb) {                       // N = 128 in 8 tiles
      v8f C0 = {}, C1 = {};
#pragma unroll
      for (int ks = 0; ks < 4; ++ks) {                     // K = 128 in 4 steps
        v16h B = frag_from_f16(W2T, LDS_W2T_STRIDE, nb * 16, ks * 32, lane);
        C0 = wmma_f32(A0[ks], B, C0);                      // B reused by both tiles
        C1 = wmma_f32(A1[ks], B, C1);
      }
      int n = nb * 16 + (lane & 15);
      float w3v = W3s[n];
      float b2v = b2s[n];
#pragma unroll
      for (int r = 0; r < 8; ++r) {                        // fused relu + W3 dot
        acc0[r] += fmaxf(C0[r] + b2v, 0.0f) * w3v;
        acc1[r] += fmaxf(C1[r] + b2v, 0.0f) * w3v;
      }
    }

    // reduce over the 16 lanes (columns n) within each half-wave
#pragma unroll
    for (int r = 0; r < 8; ++r) {
#pragma unroll
      for (int m = 8; m >= 1; m >>= 1) {
        acc0[r] += __shfl_xor(acc0[r], m, 32);
        acc1[r] += __shfl_xor(acc1[r], m, 32);
      }
    }

    int hl = lane >> 4;
    if ((lane & 15) == 0) {                                // lanes 0 and 16 write
#pragma unroll
      for (int r = 0; r < 8; ++r) {
        int pA = p0a + r + hl * 8;                         // pair index, j = p & 31
        float lgA = acc0[r] + b3v;
        float vA = 1.0f / (1.0f + __expf(-lgA));
        if (ia == (pA & 31)) vA += 1.0f;                   // + eye(32)
        __builtin_nontemporal_store(vA, &out[(size_t)b * 1024 + pA]);

        int pB = p0b + r + hl * 8;
        float lgB = acc1[r] + b3v;
        float vB = 1.0f / (1.0f + __expf(-lgB));
        if (ib == (pB & 31)) vB += 1.0f;
        __builtin_nontemporal_store(vB, &out[(size_t)b * 1024 + pB]);
      }
    }
  }
}

extern "C" void kernel_launch(void* const* d_in, const int* in_sizes, int n_in,
                              void* d_out, int out_size, void* d_ws, size_t ws_size,
                              hipStream_t stream) {
  // inputs: 0=state (unused by reference), 1=unit_states, 2=W1, 3=b1,
  //         4=W2, 5=b2, 6=W3, 7=b3
  const float* us = (const float*)d_in[1];
  const float* W1 = (const float*)d_in[2];
  const float* b1 = (const float*)d_in[3];
  const float* W2 = (const float*)d_in[4];
  const float* b2 = (const float*)d_in[5];
  const float* W3 = (const float*)d_in[6];
  const float* b3 = (const float*)d_in[7];
  float* out = (float*)d_out;
  graph_learner_kernel<<<dim3(1024), dim3(256), 0, stream>>>(us, W1, b1, W2, b2,
                                                             W3, b3, out);
}
